// gate_5892695130375
// MI455X (gfx1250) — matrix operand order repaired and checked by reading, not compiled, not run on hardware
//
#include <hip/hip_runtime.h>
#include <hip/hip_bf16.h>

#define BATCH   8
#define SEQ     2048
#define DIM     1024
#define FEAT    64
#define TOPK_N  8
#define ROWS    (BATCH*SEQ)        // 16384
#define KSTEPS  (DIM/32)           // 32 k-steps of 32
#define FRAG_ELEMS 1024            // per fragment: 512 bf16 hi + 512 bf16 lo
#define CHUNK   4                  // k-steps staged per TDM transfer
#define NCHUNKS (KSTEPS/CHUNK)     // 8
#define CHUNK_BYTES (CHUNK*12*FRAG_ELEMS*2)   // 98304 B = 96 KB
#define LDS_BYTES   (2*CHUNK_BYTES)           // 192 KB double buffer

typedef __bf16 bf16_t;
typedef unsigned int u32;
typedef __attribute__((ext_vector_type(16))) __bf16 v16bf;
typedef __attribute__((ext_vector_type(8)))  float  v8f;
typedef __attribute__((ext_vector_type(4)))  u32    v4u;
typedef __attribute__((ext_vector_type(8)))  int    v8i;
typedef __attribute__((ext_vector_type(4)))  int    v4i;

// ---------------------------------------------------------------------------
// Pre-kernel: convert f32 weights -> bf16 hi/lo fragments, pre-swizzled into
// the CDNA5 32x16 16-bit B-matrix lane layout (lane half h = lane/16, N=lane%16;
// element j of the per-lane v16bf = K 8h+j for j<8, K 16+8h+(j-8) for j>=8).
// Fragment index = (kk*3 + m)*4 + t  (12 contiguous fragments per k-step).
// ---------------------------------------------------------------------------
__global__ __launch_bounds__(256) void prep_weights_kernel(
    const float* __restrict__ Wtop, const float* __restrict__ Wfeat,
    const float* __restrict__ Wgates, bf16_t* __restrict__ frags)
{
    int id   = blockIdx.x * 256 + threadIdx.x;   // [0, 3*1024*64)
    int frag = id >> 9;                          // 512 elements per fragment
    int e    = id & 511;
    int lane = e >> 4;
    int j    = e & 15;
    int t    = frag & 3;
    int mk   = frag >> 2;
    int m    = mk % 3;
    int kk   = mk / 3;
    int K    = kk * 32 + ((lane >= 16) ? 8 : 0) + ((j < 8) ? j : 8 + j);
    int N    = t * 16 + (lane & 15);

    float w;
    if (m == 0)      w = Wtop [K * FEAT + N];
    else if (m == 1) w = Wfeat[K * FEAT + N];
    else             w = Wgates[(size_t)N * DIM + K];   // einsum "bsd,fd->bsf"

    bf16_t hi = (bf16_t)w;
    bf16_t lo = (bf16_t)(w - (float)hi);
    size_t base = (size_t)frag * FRAG_ELEMS;
    frags[base + lane * 16 + j]       = hi;
    frags[base + 512 + lane * 16 + j] = lo;
}

// ---------------------------------------------------------------------------
// TDM: 1-D DMA of one 96 KB fragment chunk (global -> LDS).  D# per ISA ch.8:
//  group0: count=1 | lds_addr | global_addr[56:0] | type=2
//  group1: data_size=8B, tile_dim0 = nbytes/8 (1-D: tile_dim1/2 = 0),
//          tensor_dim0 = 2^20 elems (>> tile, no OOB clamp), stride0 = dim0
// ---------------------------------------------------------------------------
#define TDM_TDIM0 (1u << 20)

__device__ __forceinline__ void tdm_load_chunk(u32 lds_addr, const void* gptr)
{
    unsigned long long ga = (unsigned long long)(uintptr_t)gptr;
    v4u g0;
    g0[0] = 1u;                                            // count=1 (valid)
    g0[1] = lds_addr;                                      // LDS byte address
    g0[2] = (u32)(ga & 0xFFFFFFFFu);                       // global_addr lo
    g0[3] = (u32)((ga >> 32) & 0x01FFFFFFu) | (2u << 30);  // addr[56:32]|type=2
    v8i g1;
    g1[0] = (int)(3u << 16);                               // data_size = 8 B
    g1[1] = (int)((TDM_TDIM0 & 0xFFFFu) << 16);            // tensor_dim0[15:0]
    g1[2] = (int)(TDM_TDIM0 >> 16);                        // tensor_dim0[31:16]
    g1[3] = (int)((u32)(CHUNK_BYTES / 8) << 16);           // tile_dim0 = 12288
    g1[4] = 0;                                             // tile_dim1/2 = 0
    g1[5] = (int)TDM_TDIM0;                                // dim0_stride lo32
    g1[6] = 0;
    g1[7] = 0;
    v4i z4 = {0, 0, 0, 0};
#if defined(__clang_major__) && __clang_major__ >= 23
    v8i z8 = {0, 0, 0, 0, 0, 0, 0, 0};
    __builtin_amdgcn_tensor_load_to_lds(g0, g1, z4, z4, z8, 0);
#else
    __builtin_amdgcn_tensor_load_to_lds(g0, g1, z4, z4, 0);
#endif
}

// ---------------------------------------------------------------------------
// Main fused kernel: per wave = 16 rows, all 3 GEMMs (N=64) + full epilogue.
// Weights arrive in LDS via double-buffered TDM transfers issued by wave 0.
// ---------------------------------------------------------------------------
__global__ __launch_bounds__(256) void moe_gate_kernel(
    const float* __restrict__ x,
    const float* __restrict__ b_top,
    const float* __restrict__ b_feat,
    const float* __restrict__ b_gates,
    const float* __restrict__ alpha,
    const bf16_t* __restrict__ frags,
    float* __restrict__ out)
{
    extern __shared__ __align__(128) bf16_t smem[];        // 2 x 96 KB

    const int lane = threadIdx.x & 31;
    const int wid  = threadIdx.x >> 5;                     // wave in block
    const int wave = (blockIdx.x * blockDim.x + threadIdx.x) >> 5;
    const int R    = wave * 16;          // base row of this wave's 16-row tile
    const int h    = lane >> 4;          // lane half
    const int n    = lane & 15;          // N-within-tile (B/C/D), row (A load)

    const float* xr = x + (size_t)(R + n) * DIM;
    const u32 lds_base = (u32)(uintptr_t)(void*)&smem[0];

    v8f acc[12];                         // [m*4+t]: m=0 top, m=1 feat, m=2 gates
    #pragma unroll
    for (int i = 0; i < 12; ++i) acc[i] = (v8f){0.f,0.f,0.f,0.f,0.f,0.f,0.f,0.f};

    // Prime the pipeline: chunk 0 -> buffer 0
    if (wid == 0)
        tdm_load_chunk(lds_base, frags);

    for (int c = 0; c < NCHUNKS; ++c) {
        if (wid == 0) {
            if (c + 1 < NCHUNKS) {
                tdm_load_chunk(lds_base + (u32)(((c + 1) & 1) * CHUNK_BYTES),
                               (const char*)frags + (size_t)(c + 1) * CHUNK_BYTES);
                __builtin_amdgcn_s_wait_tensorcnt(1);      // chunk c landed
            } else {
                __builtin_amdgcn_s_wait_tensorcnt(0);
            }
        }
        __syncthreads();                                   // LDS chunk c visible

        const bf16_t* buf = smem + (size_t)(c & 1) * (CHUNK_BYTES / 2);

        #pragma unroll
        for (int kc = 0; kc < CHUNK; ++kc) {
            const int kk = c * CHUNK + kc;
            // ---- A fragment: 16-bit 16x32 layout. Lane half h covers
            // K = kk*32 + h*8 + {0..7} and kk*32 + 16 + h*8 + {0..7}.
            const float* pa = xr + kk * 32 + h * 8;
            float4 q0 = *(const float4*)(pa);
            float4 q1 = *(const float4*)(pa + 4);
            float4 q2 = *(const float4*)(pa + 16);
            float4 q3 = *(const float4*)(pa + 20);
            float f[16] = { q0.x,q0.y,q0.z,q0.w, q1.x,q1.y,q1.z,q1.w,
                            q2.x,q2.y,q2.z,q2.w, q3.x,q3.y,q3.z,q3.w };
            v16bf a_hi, a_lo;
            #pragma unroll
            for (int j = 0; j < 16; ++j) {
                bf16_t hv = (bf16_t)f[j];
                a_hi[j] = hv;
                a_lo[j] = (bf16_t)(f[j] - (float)hv);
            }

            const bf16_t* kb = buf + (size_t)(kc * 12) * FRAG_ELEMS + lane * 16;
            #pragma unroll
            for (int mt = 0; mt < 12; ++mt) {
                const bf16_t* bp = kb + (size_t)mt * FRAG_ELEMS;
                v16bf b_hi = *(const v16bf*)bp;            // ds_load 32 B
                v16bf b_lo = *(const v16bf*)(bp + 512);
                // bf16x3: hi*hi + lo*hi + hi*lo ~= fp32-accurate product
                acc[mt] = __builtin_amdgcn_wmma_f32_16x16x32_bf16(
                    false, a_hi, false, b_hi, (short)0, acc[mt], false, false);
                acc[mt] = __builtin_amdgcn_wmma_f32_16x16x32_bf16(
                    false, a_lo, false, b_hi, (short)0, acc[mt], false, false);
                acc[mt] = __builtin_amdgcn_wmma_f32_16x16x32_bf16(
                    false, a_hi, false, b_lo, (short)0, acc[mt], false, false);
            }
        }
        __syncthreads();              // all reads of buf done before reuse
    }

    // ---- Epilogue: C/D layout: VGPR g, lanes 0-15 -> row g, lanes 16-31 ->
    // row 8+g, column = t*16 + n.  xor-mask reductions 1..8 stay in-half.
    const float a_mix = 1.f / (1.f + __expf(-alpha[0]));   // sigmoid(alpha)

    float bt[4], bfv[4], bg[4];
    #pragma unroll
    for (int t = 0; t < 4; ++t) {
        bt[t]  = b_top  [t * 16 + n];
        bfv[t] = b_feat [t * 16 + n];
        bg[t]  = b_gates[t * 16 + n];
    }

    #pragma unroll
    for (int g = 0; g < 8; ++g) {
        float lt[4], lf[4], gate[4];
        #pragma unroll
        for (int t = 0; t < 4; ++t) {
            lt[t] = acc[0 * 4 + t][g] + bt[t];
            lf[t] = acc[1 * 4 + t][g] + bfv[t];
            float lg = acc[2 * 4 + t][g] + bg[t];
            gate[t] = 1.f / (1.f + __expf(-lg));           // sigmoid gates
        }

        // Dense softmax branch: sum_f gate[f] * softmax(lf)[f]
        float mx = fmaxf(fmaxf(lf[0], lf[1]), fmaxf(lf[2], lf[3]));
        #pragma unroll
        for (int off = 1; off < 16; off <<= 1)
            mx = fmaxf(mx, __shfl_xor(mx, off, 32));
        float s = 0.f, gf = 0.f;
        #pragma unroll
        for (int t = 0; t < 4; ++t) {
            float e = __expf(lf[t] - mx);
            s  += e;
            gf += e * gate[t];
        }
        #pragma unroll
        for (int off = 1; off < 16; off <<= 1) {
            s  += __shfl_xor(s,  off, 32);
            gf += __shfl_xor(gf, off, 32);
        }
        float featc = gf / s;

        // Top-8 branch: iterative extract-max over the 64 router logits;
        // softmax over the 8 maxima, gather gate at each argmax by value-match.
        float w[4] = { lt[0], lt[1], lt[2], lt[3] };
        float m0 = 0.f, den = 0.f, num = 0.f;
        #pragma unroll
        for (int k = 0; k < TOPK_N; ++k) {
            float m = fmaxf(fmaxf(w[0], w[1]), fmaxf(w[2], w[3]));
            #pragma unroll
            for (int off = 1; off < 16; off <<= 1)
                m = fmaxf(m, __shfl_xor(m, off, 32));
            if (k == 0) m0 = m;
            float cg = 0.f;
            #pragma unroll
            for (int t = 0; t < 4; ++t) {
                if (w[t] == m) { cg += gate[t]; w[t] = -__builtin_inff(); }
            }
            #pragma unroll
            for (int off = 1; off < 16; off <<= 1)
                cg += __shfl_xor(cg, off, 32);
            float e = __expf(m - m0);
            den += e;
            num += e * cg;
        }
        float topc = num / den;

        float res = a_mix * topc + (1.f - a_mix) * featc;
        if (n == 0) out[R + h * 8 + g] = res;   // rows g (half 0) / 8+g (half 1)
    }
}

// ---------------------------------------------------------------------------
extern "C" void kernel_launch(void* const* d_in, const int* in_sizes, int n_in,
                              void* d_out, int out_size, void* d_ws, size_t ws_size,
                              hipStream_t stream) {
    const float* x       = (const float*)d_in[0];
    const float* W_top   = (const float*)d_in[1];
    const float* b_top   = (const float*)d_in[2];
    const float* W_feat  = (const float*)d_in[3];
    const float* b_feat  = (const float*)d_in[4];
    const float* W_gates = (const float*)d_in[5];
    const float* b_gates = (const float*)d_in[6];
    const float* alpha   = (const float*)d_in[7];
    bf16_t* frags = (bf16_t*)d_ws;              // 768 KB of hi/lo fragments
    float*  out   = (float*)d_out;

    // 3*1024*64 weight elements -> 768 blocks of 256
    prep_weights_kernel<<<(3 * DIM * FEAT) / 256, 256, 0, stream>>>(
        W_top, W_feat, W_gates, frags);

    // 16384 rows / 16 per wave = 1024 waves; 8 waves per 256-thread block;
    // 192 KB dynamic LDS for the TDM double buffer
    moe_gate_kernel<<<ROWS / 16 / 8, 256, LDS_BYTES, stream>>>(
        x, b_top, b_feat, b_gates, alpha, frags, out);
}
